// WindowAttention_31413390803641
// MI455X (gfx1250) — matrix operand order repaired and checked by reading, not compiled, not run on hardware
//
#include <hip/hip_runtime.h>
#include <hip/hip_bf16.h>

typedef __attribute__((ext_vector_type(16))) _Float16 v16h;
typedef __attribute__((ext_vector_type(8)))  _Float16 v8h;
typedef __attribute__((ext_vector_type(8)))  float    v8f;

union HV { v16h v; v8h h[2]; };

// ---------------------------------------------------------------------------
// Geometry constants for this problem instance
//   B=128, N=256 (16x16 window), C=512, H=16 heads, D=32 head dim
// ---------------------------------------------------------------------------
#define NB   128
#define NN   256
#define NC   512
#define NH   16
#define ND   32
#define NEG_INF_F (-1000000000.0f)

// CDNA5 async global->LDS DMA (tracked by ASYNCcnt, §15.18 opcode 98).
// loff: LDS byte address (low 32 bits of the generic shared pointer),
// src : 64-bit global address. 16 bytes per issue.
__device__ __forceinline__ void async_g2l_b128(unsigned loff, const void* src) {
  asm volatile("global_load_async_to_lds_b128 %0, %1, off"
               :: "v"(loff), "v"(src) : "memory");
}
__device__ __forceinline__ unsigned lds_addr(const void* p) {
  return (unsigned)(unsigned long long)p;   // addr[31:0] selects LDS offset
}

// Faithful to the reference's flat_bias.reshape(N, H, N) reinterpretation:
// bias(n,h,m) = bias_table[(n/16 - j/16 + 15), (n%16 - j%16 + 15), m%16]
// with j = h*16 + m/16.  (bias_table is (31,31,16) row-major)
__device__ __forceinline__ float rel_bias(const float* __restrict__ bt,
                                          int n, int h, int m) {
  int j  = h * 16 + (m >> 4);
  int hh = m & 15;
  int dr = (n >> 4) - (j >> 4) + 15;
  int dc = (n & 15) - (j & 15) + 15;
  return bt[(dr * 31 + dc) * 16 + hh];
}

// ---------------------------------------------------------------------------
// Kernel 1: fused QKV projection.  x(32768,512) @ {Wq,Wk,Wv}^T + bias -> f16
// stored as (b,h,n,c).  Block tile 128x64, 8 waves of 32x32, K-step 32.
// ---------------------------------------------------------------------------
__global__ __launch_bounds__(256) void qkv_gemm(
    const float* __restrict__ x,
    const float* __restrict__ Wq, const float* __restrict__ bq,
    const float* __restrict__ Wk, const float* __restrict__ bk,
    const float* __restrict__ Wv, const float* __restrict__ bv,
    _Float16* __restrict__ Qh, _Float16* __restrict__ Kh,
    _Float16* __restrict__ Vh)
{
  __shared__ __align__(16) _Float16 ldsA[128][40]; // 128 rows x 32 k (+pad)
  __shared__ __align__(16) _Float16 ldsB[64][40];  // 64 out-cols x 32 k (+pad)

  const int tid  = threadIdx.x;
  const int lane = tid & 31;
  const int wid  = tid >> 5;
  const int m0   = blockIdx.x * 128;        // global row (b*N+n) base
  const int g0   = blockIdx.y * 64;         // column in concat(Q,K,V) space
  const int mat  = g0 >> 9;                 // 0=Q 1=K 2=V (64 | 512)
  const int o0   = g0 & 511;                // column base within matrix

  const float*  Wm = (mat == 0) ? Wq : (mat == 1) ? Wk : Wv;
  const float*  bb = (mat == 0) ? bq : (mat == 1) ? bk : bv;
  _Float16*     dst = (mat == 0) ? Qh : (mat == 1) ? Kh : Vh;

  const int rb = (wid >> 1) * 32;           // wave row strip
  const int cb = (wid & 1)  * 32;           // wave col strip
  const int rlo = lane & 15;
  const int kb  = (lane >> 4) * 8;          // A-fragment k sub-offset
  const int kc  = (lane >> 4) * 16;         // unused; B fragments also use kb

  v8f acc[2][2];
  acc[0][0] = {}; acc[0][1] = {}; acc[1][0] = {}; acc[1][1] = {};

  for (int kk = 0; kk < NC; kk += 32) {
    { // stage A: 128x32 f32 -> f16
      int r = tid >> 1, seg = tid & 1;
      const float4* src = (const float4*)(x + (size_t)(m0 + r) * NC + kk + seg * 16);
      if (kk + 32 < NC) __builtin_prefetch(src + 8, 0, 1); // next k-tile
      _Float16* da = &ldsA[r][seg * 16];
      #pragma unroll
      for (int i = 0; i < 4; ++i) {
        float4 f = src[i];
        da[i*4+0] = (_Float16)f.x; da[i*4+1] = (_Float16)f.y;
        da[i*4+2] = (_Float16)f.z; da[i*4+3] = (_Float16)f.w;
      }
    }
    { // stage B: 64x32 f32 -> f16 (W is row-major (out,k), exactly B^T layout)
      int r = tid >> 2, seg = tid & 3;
      const float4* src = (const float4*)(Wm + (size_t)(o0 + r) * NC + kk + seg * 8);
      _Float16* db = &ldsB[r][seg * 8];
      #pragma unroll
      for (int i = 0; i < 2; ++i) {
        float4 f = src[i];
        db[i*4+0] = (_Float16)f.x; db[i*4+1] = (_Float16)f.y;
        db[i*4+2] = (_Float16)f.z; db[i*4+3] = (_Float16)f.w;
      }
    }
    __syncthreads();

    HV af[2], bf[2];
    #pragma unroll
    for (int mt = 0; mt < 2; ++mt) {
      const _Float16* p = &ldsA[rb + mt*16 + rlo][0];
      af[mt].h[0] = *(const v8h*)(p + kb);
      af[mt].h[1] = *(const v8h*)(p + 16 + kb);
    }
    #pragma unroll
    for (int nt = 0; nt < 2; ++nt) {
      const _Float16* p = &ldsB[cb + nt*16 + rlo][0];
      bf[nt].h[0] = *(const v8h*)(p + kb);
      bf[nt].h[1] = *(const v8h*)(p + 16 + kb);
    }
    #pragma unroll
    for (int mt = 0; mt < 2; ++mt)
      #pragma unroll
      for (int nt = 0; nt < 2; ++nt)
        acc[mt][nt] = __builtin_amdgcn_wmma_f32_16x16x32_f16(
            false, af[mt].v, false, bf[nt].v, (short)0, acc[mt][nt], false, false);
    __syncthreads();
  }

  // Epilogue: + bias, f16 store into (b,h,n,c) layout
  const int mhi = (lane >> 4) << 3;
  #pragma unroll
  for (int nt = 0; nt < 2; ++nt) {
    int o = o0 + cb + nt*16 + rlo;
    float bias = bb[o];
    int h = o >> 5, c = o & 31;
    #pragma unroll
    for (int mt = 0; mt < 2; ++mt)
      #pragma unroll
      for (int r = 0; r < 8; ++r) {
        int gm = m0 + rb + mt*16 + r + mhi;       // b*N + n
        int b = gm >> 8, n = gm & 255;
        dst[(((size_t)(b*NH + h)) * NN + n) * ND + c] =
            (_Float16)(acc[mt][nt][r] + bias);
      }
  }
}

// ---------------------------------------------------------------------------
// Kernel 2: flash-style windowed attention, one workgroup per (b,h).
// K staged by async global->LDS DMA; V^T staged manually (transpose);
// each wave owns 32 query rows; online softmax with half-wave shuffles.
// ---------------------------------------------------------------------------
__global__ __launch_bounds__(256) void attn_kernel(
    const _Float16* __restrict__ Qh, const _Float16* __restrict__ Kh,
    const _Float16* __restrict__ Vh,
    const int* __restrict__ mask, const float* __restrict__ bias_table,
    _Float16* __restrict__ AO)
{
  __shared__ __align__(16) _Float16 ldsK[256][40];   // K rows (m, c)
  __shared__ __align__(16) _Float16 ldsVt[32][264];  // V transposed (c, m)
  __shared__ __align__(16) _Float16 ldsP[8][32][40]; // per-wave P scratch

  const int tid  = threadIdx.x, lane = tid & 31, wid = tid >> 5;
  const int bh = blockIdx.x;                 // b*H + h
  const int b = bh >> 4, h = bh & 15;
  const size_t base = (size_t)bh * NN * ND;
  const float scale = 0.17677669529663687f;  // 1/sqrt(32)

  { // stage K row via async DMA: 32 halves = 4 x b128, no VGPR round-trip
    const _Float16* src = Kh + base + (size_t)tid * ND;
    unsigned loff = lds_addr(&ldsK[tid][0]);
    #pragma unroll
    for (int i = 0; i < 4; ++i)
      async_g2l_b128(loff + (unsigned)(i * 16), src + i * 8);
  }
  { // stage V transposed (needs lane scatter -> manual)
    union { v8h h[4]; _Float16 e[32]; } vr;
    const v8h* src = (const v8h*)(Vh + base + (size_t)tid * ND);
    #pragma unroll
    for (int i = 0; i < 4; ++i) vr.h[i] = src[i];
    #pragma unroll
    for (int c = 0; c < 32; ++c) ldsVt[c][tid] = vr.e[c];
  }
  asm volatile("s_wait_asynccnt 0" ::: "memory");
  __syncthreads();

  const int n0  = wid * 32;                  // this wave's query row base
  const int rlo = lane & 15;
  const int kb  = (lane >> 4) * 8;
  const int kc  = (lane >> 4) * 16;
  const int mhi = (lane >> 4) << 3;

  HV qf[2];
  #pragma unroll
  for (int mt = 0; mt < 2; ++mt) {
    const _Float16* qp = Qh + base + (size_t)(n0 + mt*16 + rlo) * ND;
    qf[mt].h[0] = *(const v8h*)(qp + kb);
    qf[mt].h[1] = *(const v8h*)(qp + 16 + kb);
  }

  float mrun[2][8], lrun[2][8];
  v8f O[2][2];
  O[0][0] = {}; O[0][1] = {}; O[1][0] = {}; O[1][1] = {};
  #pragma unroll
  for (int mt = 0; mt < 2; ++mt)
    #pragma unroll
    for (int r = 0; r < 8; ++r) { mrun[mt][r] = -1.0e30f; lrun[mt][r] = 0.0f; }

  for (int jb = 0; jb < 8; ++jb) {           // 8 key blocks of 32
    HV kf[2];
    #pragma unroll
    for (int jj = 0; jj < 2; ++jj) {
      const _Float16* kp = &ldsK[jb*32 + jj*16 + rlo][0];
      kf[jj].h[0] = *(const v8h*)(kp + kb);
      kf[jj].h[1] = *(const v8h*)(kp + 16 + kb);
    }
    v8f z = {};
    v8f s[2][2];
    #pragma unroll
    for (int mt = 0; mt < 2; ++mt)
      #pragma unroll
      for (int jj = 0; jj < 2; ++jj)
        s[mt][jj] = __builtin_amdgcn_wmma_f32_16x16x32_f16(
            false, qf[mt].v, false, kf[jj].v, (short)0, z, false, false);

    float p[2][2][8], alpha[2][8];
    #pragma unroll
    for (int mt = 0; mt < 2; ++mt) {
      #pragma unroll
      for (int r = 0; r < 8; ++r) {
        int n = n0 + mt*16 + r + mhi;
        int m0c = jb*32 + rlo;
        float v0 = s[mt][0][r] * scale + rel_bias(bias_table, n, h, m0c);
        if (mask[(n*NH + h)*NN + m0c] == 1) v0 = NEG_INF_F;
        int m1c = m0c + 16;
        float v1 = s[mt][1][r] * scale + rel_bias(bias_table, n, h, m1c);
        if (mask[(n*NH + h)*NN + m1c] == 1) v1 = NEG_INF_F;

        float bmx = fmaxf(v0, v1);           // row-max over the 32 keys
        #pragma unroll
        for (int d = 1; d < 16; d <<= 1)
          bmx = fmaxf(bmx, __shfl_xor(bmx, d, 16));
        float mnew = fmaxf(mrun[mt][r], bmx);
        float a    = __expf(mrun[mt][r] - mnew);
        float p0   = __expf(v0 - mnew);
        float p1   = __expf(v1 - mnew);
        float rs = p0 + p1;
        #pragma unroll
        for (int d = 1; d < 16; d <<= 1) rs += __shfl_xor(rs, d, 16);
        lrun[mt][r] = lrun[mt][r] * a + rs;
        mrun[mt][r] = mnew;
        alpha[mt][r] = a;
        p[mt][0][r] = p0; p[mt][1][r] = p1;
      }
      #pragma unroll
      for (int ct = 0; ct < 2; ++ct)
        #pragma unroll
        for (int r = 0; r < 8; ++r) O[mt][ct][r] *= alpha[mt][r];
    }

    // P -> per-wave LDS (C-layout -> A-layout bounce), then wait on DS counter
    #pragma unroll
    for (int mt = 0; mt < 2; ++mt)
      #pragma unroll
      for (int jj = 0; jj < 2; ++jj)
        #pragma unroll
        for (int r = 0; r < 8; ++r)
          ldsP[wid][mt*16 + r + mhi][jj*16 + rlo] = (_Float16)p[mt][jj][r];
    asm volatile("s_wait_dscnt 0" ::: "memory");

    HV pf[2], vf[2];
    #pragma unroll
    for (int mt = 0; mt < 2; ++mt) {
      const _Float16* pp = &ldsP[wid][mt*16 + rlo][0];
      pf[mt].h[0] = *(const v8h*)(pp + kb);
      pf[mt].h[1] = *(const v8h*)(pp + 16 + kb);
    }
    #pragma unroll
    for (int ct = 0; ct < 2; ++ct) {
      const _Float16* vp = &ldsVt[ct*16 + rlo][jb*32 + kb];
      vf[ct].h[0] = *(const v8h*)(vp);
      vf[ct].h[1] = *(const v8h*)(vp + 16);
    }
    #pragma unroll
    for (int mt = 0; mt < 2; ++mt)
      #pragma unroll
      for (int ct = 0; ct < 2; ++ct)
        O[mt][ct] = __builtin_amdgcn_wmma_f32_16x16x32_f16(
            false, pf[mt].v, false, vf[ct].v, (short)0, O[mt][ct], false, false);
  }

  // Normalize and store attention output as f16 (row = b*N+n, col = h*32+c)
  #pragma unroll
  for (int mt = 0; mt < 2; ++mt)
    #pragma unroll
    for (int r = 0; r < 8; ++r) {
      int n = n0 + mt*16 + r + mhi;
      float inv = 1.0f / fmaxf(lrun[mt][r], 1e-20f);
      #pragma unroll
      for (int ct = 0; ct < 2; ++ct) {
        int c = ct*16 + rlo;
        AO[((size_t)(b*NN + n)) * NC + h*ND + c] =
            (_Float16)(O[mt][ct][r] * inv);
      }
    }
}

// ---------------------------------------------------------------------------
// Kernel 3: output projection.  AO(32768,512) f16 @ Wp^T + bp -> f32 out.
// A tiles staged with async global->LDS DMA (pure f16 copy).
// ---------------------------------------------------------------------------
__global__ __launch_bounds__(256) void proj_gemm(
    const _Float16* __restrict__ AO, const float* __restrict__ Wp,
    const float* __restrict__ bp, float* __restrict__ out)
{
  __shared__ __align__(16) _Float16 ldsA[128][40];
  __shared__ __align__(16) _Float16 ldsB[64][40];

  const int tid = threadIdx.x, lane = tid & 31, wid = tid >> 5;
  const int m0 = blockIdx.x * 128;
  const int o0 = blockIdx.y * 64;
  const int rb = (wid >> 1) * 32;
  const int cb = (wid & 1)  * 32;
  const int rlo = lane & 15;
  const int kb  = (lane >> 4) * 8;
  const int kc  = (lane >> 4) * 16;

  v8f acc[2][2];
  acc[0][0] = {}; acc[0][1] = {}; acc[1][0] = {}; acc[1][1] = {};

  for (int kk = 0; kk < NC; kk += 32) {
    { // stage A (already f16) via async DMA: 16 halves = 2 x b128
      int r = tid >> 1, seg = tid & 1;
      const _Float16* src = AO + (size_t)(m0 + r) * NC + kk + seg * 16;
      if (kk + 32 < NC) __builtin_prefetch(src + 32, 0, 1);
      unsigned loff = lds_addr(&ldsA[r][seg * 16]);
      async_g2l_b128(loff,       src);
      async_g2l_b128(loff + 16u, src + 8);
    }
    { // stage B: Wp f32 -> f16
      int r = tid >> 2, seg = tid & 3;
      const float4* src = (const float4*)(Wp + (size_t)(o0 + r) * NC + kk + seg * 8);
      _Float16* db = &ldsB[r][seg * 8];
      #pragma unroll
      for (int i = 0; i < 2; ++i) {
        float4 f = src[i];
        db[i*4+0] = (_Float16)f.x; db[i*4+1] = (_Float16)f.y;
        db[i*4+2] = (_Float16)f.z; db[i*4+3] = (_Float16)f.w;
      }
    }
    asm volatile("s_wait_asynccnt 0" ::: "memory");
    __syncthreads();

    HV af[2], bf[2];
    #pragma unroll
    for (int mt = 0; mt < 2; ++mt) {
      const _Float16* p = &ldsA[rb + mt*16 + rlo][0];
      af[mt].h[0] = *(const v8h*)(p + kb);
      af[mt].h[1] = *(const v8h*)(p + 16 + kb);
    }
    #pragma unroll
    for (int nt = 0; nt < 2; ++nt) {
      const _Float16* p = &ldsB[cb + nt*16 + rlo][0];
      bf[nt].h[0] = *(const v8h*)(p + kb);
      bf[nt].h[1] = *(const v8h*)(p + 16 + kb);
    }
    #pragma unroll
    for (int mt = 0; mt < 2; ++mt)
      #pragma unroll
      for (int nt = 0; nt < 2; ++nt)
        acc[mt][nt] = __builtin_amdgcn_wmma_f32_16x16x32_f16(
            false, af[mt].v, false, bf[nt].v, (short)0, acc[mt][nt], false, false);
    __syncthreads();
  }

  const int mhi = (lane >> 4) << 3;
  #pragma unroll
  for (int nt = 0; nt < 2; ++nt) {
    int o = o0 + cb + nt*16 + rlo;
    float bias = bp[o];
    #pragma unroll
    for (int mt = 0; mt < 2; ++mt)
      #pragma unroll
      for (int r = 0; r < 8; ++r) {
        int gm = m0 + rb + mt*16 + r + mhi;
        out[(size_t)gm * NC + o] = acc[mt][nt][r] + bias;
      }
  }
}

// ---------------------------------------------------------------------------
// Launch: QKV proj -> attention -> output proj, all on `stream`.
// Workspace: Qh | Kh | Vh | AO, each 128*16*256*32 f16 = 32 MiB (128 MiB total)
// ---------------------------------------------------------------------------
extern "C" void kernel_launch(void* const* d_in, const int* in_sizes, int n_in,
                              void* d_out, int out_size, void* d_ws, size_t ws_size,
                              hipStream_t stream) {
  const float* x  = (const float*)d_in[0];
  const int*   mk = (const int*)  d_in[1];
  const float* Wq = (const float*)d_in[2];
  const float* bq = (const float*)d_in[3];
  const float* Wk = (const float*)d_in[4];
  const float* bk = (const float*)d_in[5];
  const float* Wv = (const float*)d_in[6];
  const float* bv = (const float*)d_in[7];
  const float* Wp = (const float*)d_in[8];
  const float* bp = (const float*)d_in[9];
  const float* bt = (const float*)d_in[10];
  float* out = (float*)d_out;

  char* ws = (char*)d_ws;
  const size_t SZ = (size_t)NB * NH * NN * ND * 2;  // f16 bytes per tensor
  _Float16* Qh = (_Float16*)(ws);
  _Float16* Kh = (_Float16*)(ws + SZ);
  _Float16* Vh = (_Float16*)(ws + 2 * SZ);
  _Float16* AO = (_Float16*)(ws + 3 * SZ);

  qkv_gemm<<<dim3(256, 24), 256, 0, stream>>>(x, Wq, bq, Wk, bk, Wv, bv,
                                              Qh, Kh, Vh);
  attn_kernel<<<dim3(NB * NH), 256, 0, stream>>>(Qh, Kh, Vh, mk, bt, AO);
  proj_gemm<<<dim3(256, 8), 256, 0, stream>>>(AO, Wp, bp, out);
}
